// MultiHeadAttention_23905787969980
// MI455X (gfx1250) — matrix operand order repaired and checked by reading, not compiled, not run on hardware
//
#include <hip/hip_runtime.h>
#include <hip/hip_bf16.h>

typedef __attribute__((ext_vector_type(16))) _Float16 v16h;
typedef __attribute__((ext_vector_type(8)))  _Float16 v8h;
typedef __attribute__((ext_vector_type(8)))  float    v8f;

#define L_TOK  2048
#define DMODEL 2048
#define NHEAD  16
#define DKV    128
#define LN_EPS 1e-5f
#define LDS_PITCH 40                    // 32 data halfs + 8 pad (80B rows, 16B aligned)
#define LDS_TILE  (128 * LDS_PITCH)     // one 128x32 panel
static __device__ constexpr float QK_SCALE = 0.08838834764831845f; // 1/sqrt(128)

// ---------------------------------------------------------------------------
// WMMA helper: D = A(16x32 f16) * B(32x16 f16) + C(16x16 f32)
// ---------------------------------------------------------------------------
__device__ __forceinline__ v8f wmma16(v16h a, v16h b, v8f c) {
  return __builtin_amdgcn_wmma_f32_16x16x32_f16(
      /*neg_a=*/false, a, /*neg_b=*/false, b,
      /*c_mod=*/(short)0, c, /*reuse_a=*/false, /*reuse_b=*/false);
}

// ---------------------------------------------------------------------------
// Direct-from-global fragment loaders (used by short-K and mixed GEMMs).
// A 16x32 fragment: lanes 0-15 -> row M=lane, K in {0..7,16..23};
//                   lanes 16-31 -> K in {8..15,24..31}.
// B 32x16 fragment: lanes 0-15 -> col N=lane, K in {0..7,16..23}; lanes 16-31 -> K in {8..15,24..31}.
// ---------------------------------------------------------------------------
__device__ __forceinline__ v16h load_a_frag_h(const _Float16* t, int ld) {
  const int lane = threadIdx.x & 31;
  const int r    = lane & 15;
  const int kb   = (lane >> 4) << 3;
  const _Float16* p = t + r * ld + kb;
  v8h lo = *(const v8h*)(p);
  v8h hi = *(const v8h*)(p + 16);
  v16h f;
#pragma unroll
  for (int e = 0; e < 8; ++e) { f[e] = lo[e]; f[8 + e] = hi[e]; }
  return f;
}

__device__ __forceinline__ v16h load_b_frag_h(const _Float16* t, int ld) {
  const int lane = threadIdx.x & 31;
  const int r    = lane & 15;
  const int kb   = (lane >> 4) << 3;
  const _Float16* p = t + r * ld + kb;
  v8h lo = *(const v8h*)(p);
  v8h hi = *(const v8h*)(p + 16);
  v16h f;
#pragma unroll
  for (int e = 0; e < 8; ++e) { f[e] = lo[e]; f[8 + e] = hi[e]; }
  return f;
}

__device__ __forceinline__ v16h load_a_frag_f32(const float* t, int ld) {
  const int lane = threadIdx.x & 31;
  const int r    = lane & 15;
  const int kb   = (lane >> 4) << 3;
  const float* p = t + r * ld + kb;
  float4 x0 = *(const float4*)(p);
  float4 x1 = *(const float4*)(p + 4);
  float4 x2 = *(const float4*)(p + 16);
  float4 x3 = *(const float4*)(p + 20);
  v16h f;
  f[0]=(_Float16)x0.x; f[1]=(_Float16)x0.y; f[2]=(_Float16)x0.z; f[3]=(_Float16)x0.w;
  f[4]=(_Float16)x1.x; f[5]=(_Float16)x1.y; f[6]=(_Float16)x1.z; f[7]=(_Float16)x1.w;
  f[8]=(_Float16)x2.x; f[9]=(_Float16)x2.y; f[10]=(_Float16)x2.z; f[11]=(_Float16)x2.w;
  f[12]=(_Float16)x3.x; f[13]=(_Float16)x3.y; f[14]=(_Float16)x3.z; f[15]=(_Float16)x3.w;
  return f;
}

// ---------------------------------------------------------------------------
// LDS fragment loaders (panel stored with LDS_PITCH row stride).
// ---------------------------------------------------------------------------
__device__ __forceinline__ v16h lds_a_frag(const _Float16* l) {
  const int lane = threadIdx.x & 31;
  const int r    = lane & 15;
  const int kb   = (lane >> 4) << 3;
  const _Float16* p = l + r * LDS_PITCH + kb;
  v8h lo = *(const v8h*)(p);
  v8h hi = *(const v8h*)(p + 16);
  v16h f;
#pragma unroll
  for (int e = 0; e < 8; ++e) { f[e] = lo[e]; f[8 + e] = hi[e]; }
  return f;
}

__device__ __forceinline__ v16h lds_b_frag(const _Float16* l) {
  const int lane = threadIdx.x & 31;
  const int r    = lane & 15;
  const int kb   = (lane >> 4) << 3;
  const _Float16* p = l + r * LDS_PITCH + kb;
  v8h lo = *(const v8h*)(p);
  v8h hi = *(const v8h*)(p + 16);
  v16h f;
#pragma unroll
  for (int e = 0; e < 8; ++e) { f[e] = lo[e]; f[8 + e] = hi[e]; }
  return f;
}

// ---------------------------------------------------------------------------
// Async global->LDS staging of one 128x32 f16 panel (row-major, stride ld).
// 512 16B chunks; each of the 256 threads issues 2 async copies.
// Per wave: 2 async instructions per panel (4 per A+B stage).
// ---------------------------------------------------------------------------
__device__ __forceinline__ void stage_panel(_Float16* l, const _Float16* g, int ld) {
  const int t = threadIdx.x;
#pragma unroll
  for (int c = t; c < 512; c += 256) {
    const int row = c >> 2;
    const int qq  = (c & 3) * 8;                  // 0,8,16,24 halfs
    const uint32_t loff = (uint32_t)(size_t)(l + row * LDS_PITCH + qq);
    const _Float16* gp  = g + row * ld + qq;
    asm volatile("global_load_async_to_lds_b128 %0, %1, off"
                 :: "v"(loff), "v"(gp) : "memory");
  }
}

// ---------------------------------------------------------------------------
// LDS-staged, double-buffered GEMM core (for K=2048 GEMMs).
// A rows = M (stride lda), B rows = N (stride ldb); C = A * B^T.
// ---------------------------------------------------------------------------
__device__ __forceinline__ void gemm_tile_lds(const _Float16* A, int lda,
                                              const _Float16* B, int ldb,
                                              int kTotal, int mB, int nB,
                                              int wm, int wn,
                                              _Float16* lA, _Float16* lB,
                                              v8f (&acc)[2][4]) {
  const int nsteps = kTotal >> 5;
  stage_panel(lA, A + mB * lda, lda);
  stage_panel(lB, B + nB * ldb, ldb);
  for (int i = 0; i < nsteps; ++i) {
    const int cur = i & 1;
    _Float16* cA = lA + cur * LDS_TILE;
    _Float16* cB = lB + cur * LDS_TILE;
    if (i + 1 < nsteps) {
      stage_panel(lA + (cur ^ 1) * LDS_TILE, A + mB * lda + (i + 1) * 32, lda);
      stage_panel(lB + (cur ^ 1) * LDS_TILE, B + nB * ldb + (i + 1) * 32, ldb);
      asm volatile("s_wait_asynccnt 0x4" ::: "memory");   // current buffer done
    } else {
      asm volatile("s_wait_asynccnt 0x0" ::: "memory");
    }
    __syncthreads();                                      // all waves' panels visible
    v16h a0 = lds_a_frag(cA + (wm * 32) * LDS_PITCH);
    v16h a1 = lds_a_frag(cA + (wm * 32 + 16) * LDS_PITCH);
    v16h b0 = lds_b_frag(cB + (wn * 64) * LDS_PITCH);
    v16h b1 = lds_b_frag(cB + (wn * 64 + 16) * LDS_PITCH);
    v16h b2 = lds_b_frag(cB + (wn * 64 + 32) * LDS_PITCH);
    v16h b3 = lds_b_frag(cB + (wn * 64 + 48) * LDS_PITCH);
    acc[0][0] = wmma16(a0, b0, acc[0][0]);
    acc[0][1] = wmma16(a0, b1, acc[0][1]);
    acc[0][2] = wmma16(a0, b2, acc[0][2]);
    acc[0][3] = wmma16(a0, b3, acc[0][3]);
    acc[1][0] = wmma16(a1, b0, acc[1][0]);
    acc[1][1] = wmma16(a1, b1, acc[1][1]);
    acc[1][2] = wmma16(a1, b2, acc[1][2]);
    acc[1][3] = wmma16(a1, b3, acc[1][3]);
    __syncthreads();                                      // reads done before restage
  }
}

// ---------------------------------------------------------------------------
// Direct-load GEMM cores (short-K and f32-A variants).
// ---------------------------------------------------------------------------
__device__ __forceinline__ void gemm_tile_h(const _Float16* A, int lda,
                                            const _Float16* B, int ldb,
                                            int kTotal, int m0, int n0,
                                            v8f (&acc)[2][4]) {
  for (int k = 0; k < kTotal; k += 32) {
    if (k + 32 < kTotal) {
      __builtin_prefetch(A + m0 * lda + k + 32, 0, 3);
      __builtin_prefetch(B + n0 * ldb + k + 32, 0, 3);
    }
    v16h a0 = load_a_frag_h(A + m0 * lda + k, lda);
    v16h a1 = load_a_frag_h(A + (m0 + 16) * lda + k, lda);
    v16h b0 = load_b_frag_h(B + n0 * ldb + k, ldb);
    v16h b1 = load_b_frag_h(B + (n0 + 16) * ldb + k, ldb);
    v16h b2 = load_b_frag_h(B + (n0 + 32) * ldb + k, ldb);
    v16h b3 = load_b_frag_h(B + (n0 + 48) * ldb + k, ldb);
    acc[0][0] = wmma16(a0, b0, acc[0][0]);
    acc[0][1] = wmma16(a0, b1, acc[0][1]);
    acc[0][2] = wmma16(a0, b2, acc[0][2]);
    acc[0][3] = wmma16(a0, b3, acc[0][3]);
    acc[1][0] = wmma16(a1, b0, acc[1][0]);
    acc[1][1] = wmma16(a1, b1, acc[1][1]);
    acc[1][2] = wmma16(a1, b2, acc[1][2]);
    acc[1][3] = wmma16(a1, b3, acc[1][3]);
  }
}

__device__ __forceinline__ void gemm_tile_mixed(const float* A, int lda,
                                                const _Float16* B, int ldb,
                                                int kTotal, int m0, int n0,
                                                v8f (&acc)[2][4]) {
  for (int k = 0; k < kTotal; k += 32) {
    if (k + 32 < kTotal) {
      __builtin_prefetch(A + m0 * lda + k + 32, 0, 3);
      __builtin_prefetch(B + n0 * ldb + k + 32, 0, 3);
    }
    v16h a0 = load_a_frag_f32(A + m0 * lda + k, lda);
    v16h a1 = load_a_frag_f32(A + (m0 + 16) * lda + k, lda);
    v16h b0 = load_b_frag_h(B + n0 * ldb + k, ldb);
    v16h b1 = load_b_frag_h(B + (n0 + 16) * ldb + k, ldb);
    v16h b2 = load_b_frag_h(B + (n0 + 32) * ldb + k, ldb);
    v16h b3 = load_b_frag_h(B + (n0 + 48) * ldb + k, ldb);
    acc[0][0] = wmma16(a0, b0, acc[0][0]);
    acc[0][1] = wmma16(a0, b1, acc[0][1]);
    acc[0][2] = wmma16(a0, b2, acc[0][2]);
    acc[0][3] = wmma16(a0, b3, acc[0][3]);
    acc[1][0] = wmma16(a1, b0, acc[1][0]);
    acc[1][1] = wmma16(a1, b1, acc[1][1]);
    acc[1][2] = wmma16(a1, b2, acc[1][2]);
    acc[1][3] = wmma16(a1, b3, acc[1][3]);
  }
}

// C-fragment layout: VGPR r -> (M=r, N=lane) lanes 0-15, (M=r+8, N=lane-16).
__device__ __forceinline__ void store_tile_f32(float* C, int ldc, int m0, int n0,
                                               const v8f (&acc)[2][4], float scale) {
  const int lane = threadIdx.x & 31;
  const int col  = lane & 15;
  const int rs   = (lane >> 4) * 8;
#pragma unroll
  for (int i = 0; i < 2; ++i)
#pragma unroll
    for (int j = 0; j < 4; ++j)
#pragma unroll
      for (int r = 0; r < 8; ++r)
        C[(m0 + i * 16 + rs + r) * ldc + n0 + j * 16 + col] = acc[i][j][r] * scale;
}

__device__ __forceinline__ void store_tile_h(_Float16* C, int ldc, int m0, int n0,
                                             const v8f (&acc)[2][4], float scale) {
  const int lane = threadIdx.x & 31;
  const int col  = lane & 15;
  const int rs   = (lane >> 4) * 8;
#pragma unroll
  for (int i = 0; i < 2; ++i)
#pragma unroll
    for (int j = 0; j < 4; ++j)
#pragma unroll
      for (int r = 0; r < 8; ++r)
        C[(m0 + i * 16 + rs + r) * ldc + n0 + j * 16 + col] =
            (_Float16)(acc[i][j][r] * scale);
}

__device__ __forceinline__ void store_tile_T_h(_Float16* C, int ldt, int m0, int n0,
                                               const v8f (&acc)[2][4]) {
  const int lane = threadIdx.x & 31;
  const int col  = lane & 15;
  const int rs   = (lane >> 4) * 8;
#pragma unroll
  for (int i = 0; i < 2; ++i)
#pragma unroll
    for (int j = 0; j < 4; ++j)
#pragma unroll
      for (int r = 0; r < 8; ++r)
        C[(n0 + j * 16 + col) * ldt + m0 + i * 16 + rs + r] =
            (_Float16)acc[i][j][r];
}

__device__ __forceinline__ void wave_coords(int& m0, int& n0) {
  const int w = threadIdx.x >> 5;
  m0 = blockIdx.x * 128 + (w & 3) * 32;
  n0 = blockIdx.y * 128 + (w >> 2) * 64;
}

// ---------------------------------------------------------------------------
// Kernel 0: one-shot f32 -> f16 conversion of all GEMM operands.
// ---------------------------------------------------------------------------
struct CvtArgs {
  const float* s[7];
  _Float16*    d[7];
};

__global__ __launch_bounds__(256)
void cvt_f32_f16_kernel(CvtArgs args) {
  const int z = blockIdx.y;
  const float* s = args.s[z];
  _Float16*    d = args.d[z];
  const size_t base = ((size_t)blockIdx.x * 256 + threadIdx.x) * 8;
  float4 a = *(const float4*)(s + base);
  float4 b = *(const float4*)(s + base + 4);
  v8h o;
  o[0]=(_Float16)a.x; o[1]=(_Float16)a.y; o[2]=(_Float16)a.z; o[3]=(_Float16)a.w;
  o[4]=(_Float16)b.x; o[5]=(_Float16)b.y; o[6]=(_Float16)b.z; o[7]=(_Float16)b.w;
  *(v8h*)(d + base) = o;
}

// ---------------------------------------------------------------------------
// Kernel 1: QKV projections (LDS-staged, async double-buffered).
// ---------------------------------------------------------------------------
__global__ __launch_bounds__(256)
void qkv_proj_kernel(const _Float16* __restrict__ q16, const _Float16* __restrict__ k16,
                     const _Float16* __restrict__ v16, const _Float16* __restrict__ wq16,
                     const _Float16* __restrict__ wk16, const _Float16* __restrict__ wv16,
                     _Float16* __restrict__ qh16, _Float16* __restrict__ kh16,
                     _Float16* __restrict__ vhT16) {
  __shared__ __align__(16) _Float16 lA[2 * LDS_TILE];
  __shared__ __align__(16) _Float16 lB[2 * LDS_TILE];
  const int z = blockIdx.z;
  const _Float16* A = (z == 0) ? q16 : (z == 1) ? k16 : v16;
  const _Float16* B = (z == 0) ? wq16 : (z == 1) ? wk16 : wv16;
  const int w  = threadIdx.x >> 5;
  const int wm = w & 3, wn = w >> 2;
  const int mB = blockIdx.x * 128, nB = blockIdx.y * 128;
  v8f acc[2][4] = {};
  gemm_tile_lds(A, DMODEL, B, DMODEL, DMODEL, mB, nB, wm, wn, lA, lB, acc);
  const int m0 = mB + wm * 32, n0 = nB + wn * 64;
  if (z == 0)      store_tile_h(qh16, DMODEL, m0, n0, acc, QK_SCALE);
  else if (z == 1) store_tile_h(kh16, DMODEL, m0, n0, acc, 1.0f);
  else             store_tile_T_h(vhT16, L_TOK, m0, n0, acc);
}

// ---------------------------------------------------------------------------
// Kernel 2: per-head scaled scores S[h] = Qh[h] @ Kh[h]^T (K=128), f32 out.
// ---------------------------------------------------------------------------
__global__ __launch_bounds__(256)
void attn_scores_kernel(const _Float16* __restrict__ qh16,
                        const _Float16* __restrict__ kh16,
                        float* __restrict__ attn) {
  const int h = blockIdx.z;
  int m0, n0; wave_coords(m0, n0);
  v8f acc[2][4] = {};
  gemm_tile_h(qh16 + h * DKV, DMODEL, kh16 + h * DKV, DMODEL, DKV, m0, n0, acc);
  store_tile_f32(attn + (size_t)h * L_TOK * L_TOK, L_TOK, m0, n0, acc, 1.0f);
}

// ---------------------------------------------------------------------------
// Kernel 3: in-place rowwise softmax then episodic class-weight mask.
// ---------------------------------------------------------------------------
__global__ __launch_bounds__(256)
void softmax_mask_kernel(float* __restrict__ attn, const float* __restrict__ aw,
                         const int* __restrict__ Kp) {
  const int i = blockIdx.x;
  const int h = blockIdx.y;
  float* row = attn + ((size_t)h * L_TOK + i) * L_TOK;
  __shared__ float red[256];
  const int tid = threadIdx.x;

  float vals[8];
  float m = -3.4e38f;
#pragma unroll
  for (int t = 0; t < 8; ++t) {
    vals[t] = row[tid + t * 256];
    m = fmaxf(m, vals[t]);
  }
  red[tid] = m; __syncthreads();
  for (int s = 128; s > 0; s >>= 1) {
    if (tid < s) red[tid] = fmaxf(red[tid], red[tid + s]);
    __syncthreads();
  }
  m = red[0]; __syncthreads();

  float sum = 0.f;
#pragma unroll
  for (int t = 0; t < 8; ++t) { vals[t] = __expf(vals[t] - m); sum += vals[t]; }
  red[tid] = sum; __syncthreads();
  for (int s = 128; s > 0; s >>= 1) {
    if (tid < s) red[tid] += red[tid + s];
    __syncthreads();
  }
  const float inv = 1.0f / red[0];

  const float wss = tanhf(aw[0]);
  const float wsc = tanhf(aw[1]);
  const float wdc = tanhf(aw[2]);
  const int kk = Kp[0];
  const int ci = i / kk;
#pragma unroll
  for (int t = 0; t < 8; ++t) {
    const int j = tid + t * 256;
    const float w = (j == i) ? wss : ((j / kk == ci) ? wsc : wdc);
    row[j] = vals[t] * inv * w;
  }
}

// ---------------------------------------------------------------------------
// Kernel 4: per-head O[h] = attn[h] @ V[h]; heads concatenated into o16.
// ---------------------------------------------------------------------------
__global__ __launch_bounds__(256)
void attn_v_kernel(const float* __restrict__ attn, const _Float16* __restrict__ vhT16,
                   _Float16* __restrict__ o16) {
  const int h = blockIdx.z;
  int m0, n0; wave_coords(m0, n0);   // n0 in [0,128)
  v8f acc[2][4] = {};
  gemm_tile_mixed(attn + (size_t)h * L_TOK * L_TOK, L_TOK,
                  vhT16 + h * DKV * L_TOK, L_TOK, L_TOK, m0, n0, acc);
  store_tile_h(o16 + h * DKV, DMODEL, m0, n0, acc, 1.0f);
}

// ---------------------------------------------------------------------------
// Kernel 5a: y = o @ fc_w^T + fc_b + residual (LDS-staged GEMM, f32 epilogue)
// ---------------------------------------------------------------------------
__global__ __launch_bounds__(256)
void fc_kernel(const _Float16* __restrict__ o16, const _Float16* __restrict__ fcw16,
               const float* __restrict__ fcb, const float* __restrict__ resid,
               float* __restrict__ y) {
  __shared__ __align__(16) _Float16 lA[2 * LDS_TILE];
  __shared__ __align__(16) _Float16 lB[2 * LDS_TILE];
  const int w  = threadIdx.x >> 5;
  const int wm = w & 3, wn = w >> 2;
  const int mB = blockIdx.x * 128, nB = blockIdx.y * 128;
  v8f acc[2][4] = {};
  gemm_tile_lds(o16, DMODEL, fcw16, DMODEL, DMODEL, mB, nB, wm, wn, lA, lB, acc);
  const int m0 = mB + wm * 32, n0 = nB + wn * 64;
  const int lane = threadIdx.x & 31;
  const int col  = lane & 15;
  const int rs   = (lane >> 4) * 8;
#pragma unroll
  for (int i = 0; i < 2; ++i)
#pragma unroll
    for (int j = 0; j < 4; ++j)
#pragma unroll
      for (int r = 0; r < 8; ++r) {
        const int mm = m0 + i * 16 + rs + r;
        const int nn = n0 + j * 16 + col;
        y[mm * DMODEL + nn] = acc[i][j][r] + fcb[nn] + resid[mm * DMODEL + nn];
      }
}

// ---------------------------------------------------------------------------
// Kernel 5b: rowwise LayerNorm over d_model into final output.
// ---------------------------------------------------------------------------
__global__ __launch_bounds__(256)
void layernorm_kernel(const float* __restrict__ y, const float* __restrict__ g,
                      const float* __restrict__ b, float* __restrict__ out) {
  const int row = blockIdx.x;
  const float* yr = y + (size_t)row * DMODEL;
  __shared__ float red[256];
  const int tid = threadIdx.x;

  float vals[8];
  float s = 0.f;
#pragma unroll
  for (int t = 0; t < 8; ++t) { vals[t] = yr[tid + t * 256]; s += vals[t]; }
  red[tid] = s; __syncthreads();
  for (int st = 128; st > 0; st >>= 1) {
    if (tid < st) red[tid] += red[tid + st];
    __syncthreads();
  }
  const float mu = red[0] * (1.0f / DMODEL); __syncthreads();

  float vs = 0.f;
#pragma unroll
  for (int t = 0; t < 8; ++t) { const float d = vals[t] - mu; vs += d * d; }
  red[tid] = vs; __syncthreads();
  for (int st = 128; st > 0; st >>= 1) {
    if (tid < st) red[tid] += red[tid + st];
    __syncthreads();
  }
  const float inv = rsqrtf(red[0] * (1.0f / DMODEL) + LN_EPS);
#pragma unroll
  for (int t = 0; t < 8; ++t) {
    const int j = tid + t * 256;
    out[(size_t)row * DMODEL + j] = (vals[t] - mu) * inv * g[j] + b[j];
  }
}

// ---------------------------------------------------------------------------
extern "C" void kernel_launch(void* const* d_in, const int* in_sizes, int n_in,
                              void* d_out, int out_size, void* d_ws, size_t ws_size,
                              hipStream_t stream) {
  (void)in_sizes; (void)n_in; (void)out_size; (void)ws_size;
  const float* q   = (const float*)d_in[0];
  const float* k   = (const float*)d_in[1];
  const float* v   = (const float*)d_in[2];
  const float* wqs = (const float*)d_in[3];
  const float* wks = (const float*)d_in[4];
  const float* wvs = (const float*)d_in[5];
  const float* aw  = (const float*)d_in[6];
  const float* fcw = (const float*)d_in[7];
  const float* fcb = (const float*)d_in[8];
  const float* lng = (const float*)d_in[9];
  const float* lnb = (const float*)d_in[10];
  const int*   Kp  = (const int*)d_in[12];

  float* out  = (float*)d_out;
  float* attn = out + (size_t)L_TOK * DMODEL;          // second output region

  // Workspace layout: 11 x 4M-element f16 matrices (88 MB) + one f32 (16 MB).
  const size_t MAT = (size_t)4 * 1024 * 1024;
  _Float16* h   = (_Float16*)d_ws;
  _Float16* q16   = h + 0 * MAT;
  _Float16* k16   = h + 1 * MAT;
  _Float16* v16   = h + 2 * MAT;
  _Float16* wq16  = h + 3 * MAT;
  _Float16* wk16  = h + 4 * MAT;
  _Float16* wv16  = h + 5 * MAT;
  _Float16* fcw16 = h + 6 * MAT;
  _Float16* qh16  = h + 7 * MAT;   // [L, H*DK], pre-scaled by 1/sqrt(dk)
  _Float16* kh16  = h + 8 * MAT;   // [L, H*DK]
  _Float16* vhT16 = h + 9 * MAT;   // [H*DV, L] (transposed)
  _Float16* o16   = h + 10 * MAT;  // [L, H*DV]
  float*    y     = (float*)(h + 11 * MAT);  // [L, D] f32 pre-LN

  CvtArgs ca;
  ca.s[0] = q;   ca.d[0] = q16;
  ca.s[1] = k;   ca.d[1] = k16;
  ca.s[2] = v;   ca.d[2] = v16;
  ca.s[3] = wqs; ca.d[3] = wq16;
  ca.s[4] = wks; ca.d[4] = wk16;
  ca.s[5] = wvs; ca.d[5] = wv16;
  ca.s[6] = fcw; ca.d[6] = fcw16;

  const dim3 blk(256);
  cvt_f32_f16_kernel <<<dim3(2048, 7),   blk, 0, stream>>>(ca);
  qkv_proj_kernel    <<<dim3(16, 16, 3), blk, 0, stream>>>(q16, k16, v16, wq16, wk16, wv16,
                                                           qh16, kh16, vhT16);
  attn_scores_kernel <<<dim3(16, 16, 16), blk, 0, stream>>>(qh16, kh16, attn);
  softmax_mask_kernel<<<dim3(L_TOK, NHEAD), blk, 0, stream>>>(attn, aw, Kp);
  attn_v_kernel      <<<dim3(16, 1, 16), blk, 0, stream>>>(attn, vhT16, o16);
  fc_kernel          <<<dim3(16, 16, 1), blk, 0, stream>>>(o16, fcw16, fcb, q, y);
  layernorm_kernel   <<<L_TOK, blk, 0, stream>>>(y, lng, lnb, out);
}
